// MPNN_ENN_K_Set2Set_13039520710680
// MI455X (gfx1250) — compile-verified, matrix operand order checked by reading, not run on hardware
//
#include <hip/hip_runtime.h>
#include <hip/hip_bf16.h>

// Problem constants (match reference)
constexpr int N  = 8192;
constexpr int E  = 32768;
constexpr int FN = 32;
constexpr int FE = 16;
constexpr int H  = 73;
constexpr int B  = 64;
constexpr int HH = H * H;     // 5329
constexpr int KP = 96;        // K padded to 3 x 32 for wmma f16 k=32
constexpr int NP = 5376;      // output cols padded to 21 x 256 (= 672 x 16B chunks)
constexpr int T_MP  = 3;
constexpr int T_S2S = 12;

typedef __attribute__((ext_vector_type(16))) _Float16 v16h;
typedef __attribute__((ext_vector_type(8)))  _Float16 v8h;
typedef __attribute__((ext_vector_type(8)))  float    v8f;

__device__ __forceinline__ float sigm(float x) { return 1.f / (1.f + __expf(-x)); }
__device__ __forceinline__ int   fenc(float f) { int i = __float_as_int(f); return i < 0 ? (i ^ 0x7fffffff) : i; }
__device__ __forceinline__ float fdec(int i)   { return __int_as_float(i < 0 ? (i ^ 0x7fffffff) : i); }

// ---------------- prep: W2^T -> f16 [NP, KP] (zero padded) ----------------
__global__ void k_prep_w2t(const float* __restrict__ W2, _Float16* __restrict__ w2t) {
    int idx = blockIdx.x * blockDim.x + threadIdx.x;
    if (idx >= NP * KP) return;
    int n = idx / KP, k = idx % KP;
    float v = (n < HH && k < H) ? W2[(size_t)k * HH + n] : 0.f;
    w2t[idx] = (_Float16)v;
}

// ---------------- prep: hidden = relu(ef @ W1 + b1) -> f16 [E, KP] --------
__global__ void k_hidden(const float* __restrict__ ef, const float* __restrict__ W1,
                         const float* __restrict__ b1, _Float16* __restrict__ hid) {
    __shared__ float efs[FE];
    int e = blockIdx.x, k = threadIdx.x;          // blockDim = KP = 96
    if (k < FE) efs[k] = ef[(size_t)e * FE + k];
    __syncthreads();
    float s = 0.f;
    if (k < H) {
        s = b1[k];
        #pragma unroll
        for (int f = 0; f < FE; ++f) s += efs[f] * W1[f * H + k];
        s = fmaxf(s, 0.f);
    }
    hid[(size_t)e * KP + k] = (_Float16)s;
}

// ---------------- h0 = node_feats @ W_in + b_in ---------------------------
__global__ void k_hinit(const float* __restrict__ nf, const float* __restrict__ Win,
                        const float* __restrict__ bin, float* __restrict__ h) {
    __shared__ float xs[FN];
    int n = blockIdx.x, c = threadIdx.x;          // blockDim = 96
    if (c < FN) xs[c] = nf[(size_t)n * FN + c];
    __syncthreads();
    if (c < H) {
        float s = bin[c];
        #pragma unroll
        for (int f = 0; f < FN; ++f) s += xs[f] * Win[f * H + c];
        h[(size_t)n * H + c] = s;
    }
}

// ---------------- big WMMA GEMM: A = hidden @ W2 + b2, f16 out -----------
// Wave32 tile: each wave produces a 16(M) x 64(N) strip (4 accumulators),
// K loop = 3 x 32.  A-frag: lane&15 = M row, kb = (lane>=16)?8:0, halves
// 0..7 = K kb..kb+7, halves 8..15 = K kb+16..kb+23 (ISA 7.12.2).  B-frag:
// mirrored with lane&15 = N column, sourced from pre-transposed W2 [NP,KP].
// C/D: per-lane column = lane&15, rows r(+8 for lanes>=16) across 8 VGPRs.
__global__ void k_gemm_edgeA(const _Float16* __restrict__ hid,
                             const _Float16* __restrict__ w2t,
                             const float* __restrict__ b2,
                             _Float16* __restrict__ Ah) {
    const int lane  = threadIdx.x;                // 0..31
    const int wave  = threadIdx.y;                // 0..3
    const int m0    = blockIdx.x * 16;
    const int nbase = (blockIdx.y * 16 + wave * 4) * 16;   // 4 tiles of 16 cols
    const int lrow  = lane & 15;
    const int kb    = (lane >> 4) * 8;

    v8f acc[4];
    #pragma unroll
    for (int t = 0; t < 4; ++t) {
        int col = nbase + t * 16 + lrow;
        float bias = (col < HH) ? b2[col] : 0.f;
        #pragma unroll
        for (int r = 0; r < 8; ++r) acc[t][r] = bias;
    }

    #pragma unroll
    for (int kk = 0; kk < KP; kk += 32) {
        const _Float16* ap = hid + (size_t)(m0 + lrow) * KP + kk + kb;
        v8h alo = *(const v8h*)ap;
        v8h ahi = *(const v8h*)(ap + 16);
        v16h a;
        #pragma unroll
        for (int i = 0; i < 8; ++i) { a[i] = alo[i]; a[8 + i] = ahi[i]; }

        #pragma unroll
        for (int t = 0; t < 4; ++t) {
            const _Float16* bp = w2t + (size_t)(nbase + t * 16 + lrow) * KP + kk + kb;
            v8h blo = *(const v8h*)bp;
            v8h bhi = *(const v8h*)(bp + 16);
            v16h bm;
            #pragma unroll
            for (int i = 0; i < 8; ++i) { bm[i] = blo[i]; bm[8 + i] = bhi[i]; }
            acc[t] = __builtin_amdgcn_wmma_f32_16x16x32_f16(
                false, a, false, bm, (short)0, acc[t], false, false);
        }
    }

    const int rbase = m0 + ((lane >> 4) << 3);
    #pragma unroll
    for (int t = 0; t < 4; ++t) {
        int col = nbase + t * 16 + lrow;
        #pragma unroll
        for (int r = 0; r < 8; ++r)
            Ah[(size_t)(rbase + r) * NP + col] = (_Float16)acc[t][r];
    }
}

// ---------------- generic zero ------------------------------------------
__global__ void k_zero(float* __restrict__ p, int n) {
    int i = blockIdx.x * blockDim.x + threadIdx.x;
    if (i < n) p[i] = 0.f;
}

// ---------------- per-edge matvec + scatter-add (memory bound on Ah) -----
// Stage the full padded A row (10.75 KB) into LDS with coalesced b128
// copies (row stride NP*2 = 672 x 16B, perfectly aligned), then compute the
// 73 dot products out of LDS.  This turns the dominant HBM stream (3 x
// 352 MB over the MP steps) into dense 128B-line traffic with ~8x fewer
// VMEM requests than per-element u16 loads.
__global__ void k_edgemsg(const _Float16* __restrict__ Ah, const float* __restrict__ h,
                          const int* __restrict__ Esrc, const int* __restrict__ Etgt,
                          float* __restrict__ agg) {
    __shared__ _Float16 arow[NP];                 // 10752 B
    __shared__ float hs[H];
    const int e = blockIdx.x;
    const int t = threadIdx.x;                    // blockDim = 128
    const int src = Esrc[e], tgt = Etgt[e];
    if (t < H) hs[t] = h[(size_t)src * H + t];

    const uint4* gsrc = (const uint4*)(Ah + (size_t)e * NP);
    uint4* ldst = (uint4*)arow;
    #pragma unroll
    for (int c = t; c < NP / 8; c += 128)         // 672 chunks of 16B
        ldst[c] = gsrc[c];
    __syncthreads();

    if (t < H) {
        const _Float16* ar = arow + t * H;
        float s = 0.f;
        #pragma unroll 8
        for (int j = 0; j < H; ++j) s += (float)ar[j] * hs[j];
        atomicAdd(&agg[(size_t)tgt * H + t], s);
    }
}

// ---------------- GRU cell update (per node, in place) -------------------
__global__ void k_gru(const float* __restrict__ agg, const float* __restrict__ Wih,
                      const float* __restrict__ Whh, const float* __restrict__ bih,
                      const float* __restrict__ bhh, float* __restrict__ h) {
    __shared__ float as[H], hsv[H];
    int n = blockIdx.x, c = threadIdx.x;          // blockDim = 96
    if (c < H) { as[c] = agg[(size_t)n * H + c]; hsv[c] = h[(size_t)n * H + c]; }
    __syncthreads();
    if (c < H) {
        float ir = bih[c], iz = bih[H + c], in_ = bih[2 * H + c];
        float hr = bhh[c], hz = bhh[H + c], hn  = bhh[2 * H + c];
        const float* wr = Wih + (size_t)c * H;
        const float* wz = Wih + (size_t)(H + c) * H;
        const float* wn = Wih + (size_t)(2 * H + c) * H;
        const float* ur = Whh + (size_t)c * H;
        const float* uz = Whh + (size_t)(H + c) * H;
        const float* un = Whh + (size_t)(2 * H + c) * H;
        for (int k = 0; k < H; ++k) {
            float a = as[k], hv = hsv[k];
            ir += wr[k] * a;  iz += wz[k] * a;  in_ += wn[k] * a;
            hr += ur[k] * hv; hz += uz[k] * hv; hn  += un[k] * hv;
        }
        float r  = sigm(ir + hr);
        float z  = sigm(iz + hz);
        float nn = tanhf(in_ + r * hn);
        h[(size_t)n * H + c] = (1.f - z) * nn + z * hsv[c];
    }
}

// ---------------- Set2Set: attention scratch init ------------------------
__global__ void k_attn_init(int* __restrict__ emaxi, float* __restrict__ asum,
                            float* __restrict__ rbuf) {
    int i = blockIdx.x * blockDim.x + threadIdx.x;
    if (i < B) { emaxi[i] = (int)0x80000000; asum[i] = 0.f; }
    if (i < B * H) rbuf[i] = 0.f;
}

// ---------------- Set2Set: LSTM cell (one block per batch element) -------
__global__ void k_lstm(const float* __restrict__ qstar, const float* __restrict__ Wih,
                       const float* __restrict__ Whh, const float* __restrict__ bih,
                       const float* __restrict__ bhh, float* __restrict__ hh,
                       float* __restrict__ cc) {
    __shared__ float qs[2 * H], hs2[H], gs[4 * H];
    int b = blockIdx.x, t = threadIdx.x;          // blockDim = 320
    if (t < 2 * H) qs[t]  = qstar[(size_t)b * 2 * H + t];
    if (t < H)     hs2[t] = hh[(size_t)b * H + t];
    __syncthreads();
    if (t < 4 * H) {
        float g = bih[t] + bhh[t];
        const float* wi = Wih + (size_t)t * (2 * H);
        const float* wh = Whh + (size_t)t * H;
        for (int k = 0; k < 2 * H; ++k) g += wi[k] * qs[k];
        for (int k = 0; k < H; ++k)     g += wh[k] * hs2[k];
        gs[t] = g;
    }
    __syncthreads();
    if (t < H) {
        float ig = gs[t], fg = gs[H + t], gg = gs[2 * H + t], og = gs[3 * H + t];
        float c_ = sigm(fg) * cc[(size_t)b * H + t] + sigm(ig) * tanhf(gg);
        float h_ = sigm(og) * tanhf(c_);
        cc[(size_t)b * H + t] = c_;
        hh[(size_t)b * H + t] = h_;
    }
}

// ---------------- Set2Set: attention logits + segment max ----------------
__global__ void k_emax(const float* __restrict__ h, const float* __restrict__ hh,
                       const int* __restrict__ batch, float* __restrict__ evec,
                       int* __restrict__ emaxi) {
    int n = blockIdx.x * 8 + (threadIdx.x >> 5);  // wave per node, blockDim = 256
    int l = threadIdx.x & 31;
    int b = batch[n];
    const float* hr = h  + (size_t)n * H;
    const float* qr = hh + (size_t)b * H;
    float s = 0.f;
    for (int j = l; j < H; j += 32) s += hr[j] * qr[j];
    #pragma unroll
    for (int off = 16; off > 0; off >>= 1) s += __shfl_xor(s, off, 32);
    if (l == 0) { evec[n] = s; atomicMax(&emaxi[b], fenc(s)); }
}

// ---------------- Set2Set: exp + segment sum -----------------------------
__global__ void k_soft(const float* __restrict__ evec, const int* __restrict__ emaxi,
                       const int* __restrict__ batch, float* __restrict__ avec,
                       float* __restrict__ asum) {
    int n = blockIdx.x * blockDim.x + threadIdx.x;
    if (n < N) {
        int b = batch[n];
        float a = __expf(evec[n] - fdec(emaxi[b]));
        avec[n] = a;
        atomicAdd(&asum[b], a);
    }
}

// ---------------- Set2Set: weighted readout r ----------------------------
__global__ void k_racc(const float* __restrict__ h, const float* __restrict__ avec,
                       const float* __restrict__ asum, const int* __restrict__ batch,
                       float* __restrict__ rbuf) {
    int n = blockIdx.x, i = threadIdx.x;          // blockDim = 96
    if (i < H) {
        int b = batch[n];
        float w = avec[n] / asum[b];
        atomicAdd(&rbuf[(size_t)b * H + i], w * h[(size_t)n * H + i]);
    }
}

// ---------------- Set2Set: q_star = [q, r] -------------------------------
__global__ void k_qstar(const float* __restrict__ hh, const float* __restrict__ rbuf,
                        float* __restrict__ qstar) {
    int i = blockIdx.x * blockDim.x + threadIdx.x;
    if (i < B * 2 * H) {
        int b = i / (2 * H), t = i % (2 * H);
        qstar[i] = (t < H) ? hh[(size_t)b * H + t] : rbuf[(size_t)b * H + (t - H)];
    }
}

// ---------------- output linear ------------------------------------------
__global__ void k_out(const float* __restrict__ hh, const float* __restrict__ Wout,
                      const float* __restrict__ bout, float* __restrict__ out) {
    int b = threadIdx.x;                          // blockDim = 64
    if (b < B) {
        float s = bout[0];
        for (int c = 0; c < H; ++c) s += hh[(size_t)b * H + c] * Wout[c];
        out[b] = s;
    }
}

extern "C" void kernel_launch(void* const* d_in, const int* in_sizes, int n_in,
                              void* d_out, int out_size, void* d_ws, size_t ws_size,
                              hipStream_t stream) {
    (void)in_sizes; (void)n_in; (void)out_size; (void)ws_size;
    const float* node_feats = (const float*)d_in[0];
    const float* edge_feats = (const float*)d_in[1];
    const float* W_in   = (const float*)d_in[2];
    const float* b_in   = (const float*)d_in[3];
    const float* ee_W1  = (const float*)d_in[4];
    const float* ee_b1  = (const float*)d_in[5];
    const float* ee_W2  = (const float*)d_in[6];
    const float* ee_b2  = (const float*)d_in[7];
    const float* gru_Wih = (const float*)d_in[8];
    const float* gru_Whh = (const float*)d_in[9];
    const float* gru_bih = (const float*)d_in[10];
    const float* gru_bhh = (const float*)d_in[11];
    const float* lstm_Wih = (const float*)d_in[12];
    const float* lstm_Whh = (const float*)d_in[13];
    const float* lstm_bih = (const float*)d_in[14];
    const float* lstm_bhh = (const float*)d_in[15];
    const float* W_out = (const float*)d_in[16];
    const float* b_out = (const float*)d_in[17];
    const int* Esrc  = (const int*)d_in[18];
    const int* Etgt  = (const int*)d_in[19];
    const int* batch = (const int*)d_in[20];

    // Carve workspace (dominant: Ah = E*NP f16 = 352 MB)
    char* p = (char*)d_ws;
    auto carve = [&](size_t bytes) -> void* {
        void* r = (void*)p;
        p += (bytes + 255) & ~(size_t)255;
        return r;
    };
    _Float16* hid  = (_Float16*)carve((size_t)E * KP * sizeof(_Float16));
    _Float16* w2t  = (_Float16*)carve((size_t)NP * KP * sizeof(_Float16));
    _Float16* Ah   = (_Float16*)carve((size_t)E * NP * sizeof(_Float16));
    float* h     = (float*)carve((size_t)N * H * sizeof(float));
    float* agg   = (float*)carve((size_t)N * H * sizeof(float));
    float* qstar = (float*)carve((size_t)B * 2 * H * sizeof(float));
    float* hh    = (float*)carve((size_t)B * H * sizeof(float));
    float* cc    = (float*)carve((size_t)B * H * sizeof(float));
    float* rbuf  = (float*)carve((size_t)B * H * sizeof(float));
    float* evec  = (float*)carve((size_t)N * sizeof(float));
    float* avec  = (float*)carve((size_t)N * sizeof(float));
    int*   emaxi = (int*)carve((size_t)B * sizeof(int));
    float* asum  = (float*)carve((size_t)B * sizeof(float));

    // --- Edge encoder: hidden, transposed W2, then the big WMMA GEMM -> A (f16)
    k_prep_w2t<<<(NP * KP + 255) / 256, 256, 0, stream>>>(ee_W2, w2t);
    k_hidden<<<E, KP, 0, stream>>>(edge_feats, ee_W1, ee_b1, hid);
    k_hinit<<<N, 96, 0, stream>>>(node_feats, W_in, b_in, h);
    k_gemm_edgeA<<<dim3(E / 16, NP / 256), dim3(32, 4), 0, stream>>>(hid, w2t, ee_b2, Ah);

    // --- Message passing: 3 x (matvec-scatter + GRU)
    for (int s = 0; s < T_MP; ++s) {
        k_zero<<<(N * H + 255) / 256, 256, 0, stream>>>(agg, N * H);
        k_edgemsg<<<E, 128, 0, stream>>>(Ah, h, Esrc, Etgt, agg);
        k_gru<<<N, 96, 0, stream>>>(agg, gru_Wih, gru_Whh, gru_bih, gru_bhh, h);
    }

    // --- Set2Set pooling
    k_zero<<<(B * 2 * H + 255) / 256, 256, 0, stream>>>(qstar, B * 2 * H);
    k_zero<<<(B * H + 255) / 256, 256, 0, stream>>>(hh, B * H);
    k_zero<<<(B * H + 255) / 256, 256, 0, stream>>>(cc, B * H);
    for (int s = 0; s < T_S2S; ++s) {
        k_lstm<<<B, 320, 0, stream>>>(qstar, lstm_Wih, lstm_Whh, lstm_bih, lstm_bhh, hh, cc);
        k_attn_init<<<(B * H + 255) / 256, 256, 0, stream>>>(emaxi, asum, rbuf);
        k_emax<<<N / 8, 256, 0, stream>>>(h, hh, batch, evec, emaxi);
        k_soft<<<(N + 255) / 256, 256, 0, stream>>>(evec, emaxi, batch, avec, asum);
        k_racc<<<N, 96, 0, stream>>>(h, avec, asum, batch, rbuf);
        k_qstar<<<(B * 2 * H + 255) / 256, 256, 0, stream>>>(hh, rbuf, qstar);
    }

    // --- Output linear (uses q part = hh after last step)
    k_out<<<1, 64, 0, stream>>>(hh, W_out, b_out, (float*)d_out);
}
